// ContinuousPositionBias_55937654063662
// MI455X (gfx1250) — matrix operand order repaired and checked by reading, not compiled, not run on hardware
//
#include <hip/hip_runtime.h>
#include <hip/hip_bf16.h>
#include <math.h>

// ContinuousPositionBias for MI455X (gfx1250, wave32).
// Stage 1: fused MLP  bias[b,t,h] = relu(table@W1+b1) @ W2  via v_wmma_f32_16x16x32_f16
// Stage 2: gather through rpi + zero-prefix + transpose, NT stores (682 MB stream).

typedef __attribute__((ext_vector_type(16))) _Float16 v16h;
typedef __attribute__((ext_vector_type(8)))  float    v8f;

#define CPB_NTAB      2209   // (2*24-1)^2
#define CPB_NTAB_PAD  2240   // 140 tiles * 16 rows: lets WMMA stores be unguarded
#define CPB_KD        512    // RPB_DIM
#define CPB_NH        16     // NUM_HEADS
#define CPB_N         576    // 24*24
#define CPB_NP        577    // N + num_prefix_tokens(=1)
#define CPB_BS        32
#define CPB_TILES     139    // ceil(2209/16)
#define W2_STRIDE     520    // 512 + 8 f16 pad (16B) -> n-rows land on distinct LDS banks

__global__ __launch_bounds__(128) void cpb_bias_wmma_kernel(
    const float* __restrict__ glob_pos,   // (32,4)
    const float* __restrict__ ctab,       // (2209,2)
    const float* __restrict__ W1,         // (2,512)
    const float* __restrict__ b1,         // (512,)
    const float* __restrict__ W2,         // (512,16)
    float* __restrict__ bias)             // (32, 2240, 16) scratch (padded rows)
{
  __shared__ float sW1x[CPB_KD];
  __shared__ float sW1y[CPB_KD];
  __shared__ float sB1 [CPB_KD];
  // W2 transposed to [n][k] (f16) so a lane's 16 consecutive-K values are one 32B read.
  __shared__ __attribute__((aligned(32))) _Float16 sW2t[CPB_NH * W2_STRIDE];

  const int tid = threadIdx.x;
  for (int k = tid; k < CPB_KD; k += 128) {
    sW1x[k] = W1[k];
    sW1y[k] = W1[CPB_KD + k];
    sB1[k]  = b1[k];
  }
  for (int e = tid; e < CPB_KD * CPB_NH; e += 128) {
    const int k = e >> 4;        // W2 is row-major (k, n)
    const int n = e & 15;
    sW2t[n * W2_STRIDE + k] = (_Float16)W2[e];
  }
  __syncthreads();

  const int b    = blockIdx.y;
  const int tile = blockIdx.x * 4 + (tid >> 5);   // one 16-row tile per wave
  if (tile >= CPB_TILES + 1) return;              // 35*4=140 tiles; tile 139 is pad-only

  const int lane = tid & 31;
  const int m    = lane & 15;   // A-matrix row owned by this lane; also B/C column N
  const int half = lane >> 4;

  // Per-batch log-CPB offset (reference: pos = g[2:]/g[:2]*8 -> signed log2/3 -> *2-1)
  const float g0v = glob_pos[b * 4 + 0];
  const float g1v = glob_pos[b * 4 + 1];
  const float g2v = glob_pos[b * 4 + 2];
  const float g3v = glob_pos[b * 4 + 3];
  float p0 = g2v / g0v * 8.0f;
  float p1 = g3v / g1v * 8.0f;
  p0 = copysignf(log2f(fabsf(p0) + 1.0f) * (1.0f / 3.0f), p0) * 2.0f - 1.0f;
  p1 = copysignf(log2f(fabsf(p1) + 1.0f) * (1.0f / 3.0f), p1) * 2.0f - 1.0f;

  int tr = tile * 16 + m;
  if (tr > CPB_NTAB - 1) tr = CPB_NTAB - 1;       // clamp: pad rows duplicate last row
  const float tx = ctab[tr * 2 + 0] + p0;
  const float ty = ctab[tr * 2 + 1] + p1;

  v8f c = {};
  const _Float16* w2row = &sW2t[m * W2_STRIDE];

#pragma unroll 4
  for (int k0 = 0; k0 < CPB_KD; k0 += 32) {
    // A operand, 16x32 f16, contiguous k order used here (matches B):
    //   lane half h, element j<8  -> K = k0 + 16h + j
    //   element 8+j               -> K = k0 + 16h + 8 + j
    const int ga = k0 + 16 * half;
    const int gb = k0 + 16 * half + 8;
    v16h a;
#pragma unroll
    for (int j = 0; j < 8; ++j) {
      float h0 = fmaxf(tx * sW1x[ga + j] + ty * sW1y[ga + j] + sB1[ga + j], 0.0f);
      float h1 = fmaxf(tx * sW1x[gb + j] + ty * sW1y[gb + j] + sB1[gb + j], 0.0f);
      a[j]     = (_Float16)h0;
      a[8 + j] = (_Float16)h1;
    }
    // B operand, 32x16 f16: lanes 0-15 hold K = k0+0..15, lanes 16-31 hold K = k0+16..31,
    // N = lane&15. One aligned 32-byte LDS read.
    const v16h bb = *(const v16h*)(w2row + k0 + 16 * half);

    c = __builtin_amdgcn_wmma_f32_16x16x32_f16(
        /*neg_a=*/false, a, /*neg_b=*/false, bb,
        /*c_mod=*/(short)0, c, /*reuse_a=*/false, /*reuse_b=*/false);
  }

  // C/D layout: VGPR r -> M = r + 8*half, N = lane&15.
  // Buffer is row-padded to 2240, so no per-row guard: straight coalesced stores.
  float* dst = bias + ((size_t)b * CPB_NTAB_PAD + (size_t)tile * 16 + 8 * half) * CPB_NH + m;
#pragma unroll
  for (int r = 0; r < 8; ++r)
    dst[r * CPB_NH] = c[r];
}

__global__ __launch_bounds__(256) void cpb_scatter_kernel(
    const int*   __restrict__ rpi,        // (576,576)
    const float* __restrict__ bias,       // (32, 2240, 16)
    const int*   __restrict__ p_prefix,   // scalar (==1)
    float*       __restrict__ out)        // (32,16,577,577)
{
  const int ij = blockIdx.x * 256 + threadIdx.x;   // flattened (i,j) within one batch
  if (ij >= CPB_NP * CPB_NP) return;
  const int b = blockIdx.y;
  const int i = ij / CPB_NP;                       // const-div -> mul+shift
  const int j = ij - i * CPB_NP;
  const int P = *p_prefix;

  float v[CPB_NH];
  if (i < P || j < P) {
#pragma unroll
    for (int h = 0; h < CPB_NH; ++h) v[h] = 0.0f;
  } else {
    const int idx = rpi[(i - P) * CPB_N + (j - P)];
    const float4* bp =
        (const float4*)(bias + ((size_t)b * CPB_NTAB_PAD + (size_t)idx) * CPB_NH);
    const float4 q0 = bp[0], q1 = bp[1], q2 = bp[2], q3 = bp[3];
    v[0]=q0.x;  v[1]=q0.y;  v[2]=q0.z;  v[3]=q0.w;
    v[4]=q1.x;  v[5]=q1.y;  v[6]=q1.z;  v[7]=q1.w;
    v[8]=q2.x;  v[9]=q2.y;  v[10]=q2.z; v[11]=q2.w;
    v[12]=q3.x; v[13]=q3.y; v[14]=q3.z; v[15]=q3.w;
  }

  const size_t plane = (size_t)CPB_NP * CPB_NP;
  float* o = out + (size_t)b * CPB_NH * plane + (size_t)ij;
  // 682 MB streaming output: non-temporal so it doesn't rinse rpi/bias out of L2.
#pragma unroll
  for (int h = 0; h < CPB_NH; ++h)
    __builtin_nontemporal_store(v[h], o + (size_t)h * plane);
}

extern "C" void kernel_launch(void* const* d_in, const int* in_sizes, int n_in,
                              void* d_out, int out_size, void* d_ws, size_t ws_size,
                              hipStream_t stream) {
  (void)in_sizes; (void)n_in; (void)out_size; (void)ws_size;
  const float* glob_pos = (const float*)d_in[0];   // (1,32,4)
  const float* ctab     = (const float*)d_in[1];   // (2209,2)
  const int*   rpi      = (const int*)  d_in[2];   // (576,576)
  const float* W1       = (const float*)d_in[3];   // (2,512)
  const float* b1       = (const float*)d_in[4];   // (512,)
  const float* W2       = (const float*)d_in[5];   // (512,16)
  const int*   pfx      = (const int*)  d_in[6];   // scalar 1

  float* bias = (float*)d_ws;                      // needs 32*2240*16*4 = 4.59 MB
  float* out  = (float*)d_out;                     // (32,16,577,577)

  // Stage 1: 140 tiles (139 real + 1 pad) of 16 table rows, 4 waves per block, 32 batches.
  dim3 g1(35, CPB_BS, 1);
  cpb_bias_wmma_kernel<<<g1, 128, 0, stream>>>(glob_pos, ctab, W1, b1, W2, bias);

  // Stage 2: one thread per (b, flattened i*577+j), 16 heads per thread.
  dim3 g2((CPB_NP * CPB_NP + 255) / 256, CPB_BS, 1);
  cpb_scatter_kernel<<<g2, 256, 0, stream>>>(rpi, bias, pfx, out);
}
